// SNN_small_60120952210053
// MI455X (gfx1250) — matrix operand order repaired and checked by reading, not compiled, not run on hardware
//
#include <hip/hip_runtime.h>
#include <hip/hip_bf16.h>

typedef _Float16 v16h __attribute__((ext_vector_type(16)));
typedef _Float16 v8h  __attribute__((ext_vector_type(8)));
typedef float    v8f  __attribute__((ext_vector_type(8)));
typedef float    v4f  __attribute__((ext_vector_type(4)));
typedef unsigned int u32x4 __attribute__((ext_vector_type(4)));
typedef int      i32x4 __attribute__((ext_vector_type(4)));
typedef int      i32x8 __attribute__((ext_vector_type(8)));

#define SNN_B    4096
#define SNN_T    128
#define SNN_DIN  64
#define SNN_H    64
#define SNN_DOUT 16
#define SNN_BETA 0.9f
#define SNN_THR  1.0f

// A-fragment from f16 LDS: 16 contiguous halves along K.
__device__ __forceinline__ v16h fragA_f16(const _Float16* p) {
    v8h lo = *(const v8h*)p;
    v8h hi = *(const v8h*)(p + 8);
    return __builtin_shufflevector(lo, hi, 0,1,2,3,4,5,6,7,8,9,10,11,12,13,14,15);
}

// A-fragment from f32 LDS tile: 16 contiguous floats at p, convert to f16.
__device__ __forceinline__ v16h fragA_f32(const float* p) {
    v4f a0 = *(const v4f*)(p);
    v4f a1 = *(const v4f*)(p + 4);
    v4f b0 = *(const v4f*)(p + 8);
    v4f b1 = *(const v4f*)(p + 12);
    v16h h;
    #pragma unroll
    for (int e = 0; e < 4; e++) {
        h[e]      = (_Float16)a0[e];
        h[4 + e]  = (_Float16)a1[e];
        h[8 + e]  = (_Float16)b0[e];
        h[12 + e] = (_Float16)b1[e];
    }
    return h;
}

// B-fragment (this code's contiguous-K order, same as the A fragments): 16 contiguous halves along K.
__device__ __forceinline__ v16h fragB_f16(const _Float16* p) {
    v8h lo = *(const v8h*)p;
    v8h hi = *(const v8h*)(p + 8);
    return __builtin_shufflevector(lo, hi, 0,1,2,3,4,5,6,7,8,9,10,11,12,13,14,15);
}

// TDM: DMA a 16x64 f32 tile (row stride T*DIN elements) from global to LDS.
__device__ __forceinline__ void tdm_load_x_tile(const float* gsrc, unsigned lds_off) {
    unsigned long long ga = (unsigned long long)(uintptr_t)gsrc;
    u32x4 g0;
    g0[0] = 1u;                                                  // count=1 user D#
    g0[1] = lds_off;                                             // lds_addr (bytes)
    g0[2] = (unsigned)(ga & 0xFFFFFFFFu);                        // global_addr lo
    g0[3] = (unsigned)((ga >> 32) & 0x01FFFFFFull) | (2u << 30); // global_addr hi | type=2
    i32x8 g1;
    g1[0] = (int)(2u << 16);                 // data_size = 4 bytes
    g1[1] = (int)((unsigned)SNN_DIN << 16);  // tensor_dim0 = 64
    g1[2] = (int)(16u << 16);                // tensor_dim1 = 16 (dim0 hi16 = 0)
    g1[3] = (int)((unsigned)SNN_DIN << 16);  // tile_dim0 = 64 (dim1 hi16 = 0)
    g1[4] = 16;                              // tile_dim1 = 16, tile_dim2 = 0
    g1[5] = SNN_T * SNN_DIN;                 // tensor_dim0_stride = 8192 elements
    g1[6] = 0;
    g1[7] = 0;
    i32x4 z4 = {};
#if __has_include(<hip/amd_detail/amd_gfx1250_TDM.h>)
    i32x8 z8 = {};
    __builtin_amdgcn_tensor_load_to_lds(g0, g1, z4, z4, z8, 0);
#else
    __builtin_amdgcn_tensor_load_to_lds(g0, g1, z4, z4, 0);
#endif
}

__global__ __launch_bounds__(128)
void snn_small_kernel(const float* __restrict__ x,  const float* __restrict__ W1,
                      const float* __restrict__ b1, const float* __restrict__ W2,
                      const float* __restrict__ b2, const float* __restrict__ Wr,
                      const float* __restrict__ br, float* __restrict__ out)
{
    __shared__ __attribute__((aligned(16))) _Float16 sW1[SNN_H * SNN_DIN];  // 8KB
    __shared__ __attribute__((aligned(16))) _Float16 sW2[SNN_H * SNN_H];    // 8KB
    __shared__ __attribute__((aligned(16))) float    sXf[2][16 * SNN_DIN];  // 2x4KB x tiles (TDM dest)
    __shared__ __attribute__((aligned(16))) _Float16 sS[16 * SNN_H];        // 2KB spike exchange

    const int tid  = threadIdx.x;
    const int w    = tid >> 5;        // wave id -> owns n-tile j = w (h = 16w..16w+15)
    const int lane = tid & 31;
    const int lo16 = lane & 15;
    const int hi   = lane >> 4;       // 0 or 1
    const int b0   = blockIdx.x * 16; // batch tile base
    const float* xbase = x + (size_t)b0 * SNN_T * SNN_DIN;

    // Kick off TDM for x(t=0) immediately (wave 0 issues; EXEC-independent DMA).
    if (w == 0)
        tdm_load_x_tile(xbase, (unsigned)(uintptr_t)&sXf[0][0]);

    // Stage weights to LDS as f16 (once).
    for (int i = tid; i < SNN_H * SNN_DIN; i += 128) sW1[i] = (_Float16)W1[i];
    for (int i = tid; i < SNN_H * SNN_H;   i += 128) sW2[i] = (_Float16)W2[i];
    __syncthreads();

    // Hoisted weight B-fragments for this wave's n-tile (K=64 -> chunks c=0,1).
    const int brow = 16 * w + lo16;   // B lane n -> weight row (h)
    const int kb   = 16 * hi;         // B lane K base within 32-chunk
    const v16h bw1_0 = fragB_f16(&sW1[brow * SNN_DIN + 0  + kb]);
    const v16h bw1_1 = fragB_f16(&sW1[brow * SNN_DIN + 32 + kb]);
    const v16h bw2_0 = fragB_f16(&sW2[brow * SNN_H   + 0  + kb]);
    const v16h bw2_1 = fragB_f16(&sW2[brow * SNN_H   + 32 + kb]);
    const float b1v = b1[brow];
    const float b2v = b2[brow];

    v8f mem1 = {};
    v8f mem2 = {};
    v8f ssum = {};

    float* outR = out;                          // readout_spk [B,16]
    float* outS = out + SNN_B * SNN_DOUT;       // spk2 record [B,T,H]

    const int hcol = 16 * w + lo16;             // D-layout n (= h) for this lane
    const int koff = 16 * hi;                   // A-layout K offset
    const int xr   = tid >> 3;                  // prefetch helper row
    const int xc   = (tid & 7) * 8;             // prefetch helper col

    for (int t = 0; t < SNN_T; ++t) {
        const int cur = t & 1;

        // Issue TDM for x(t+1) into the other buffer, then wait for x(t).
        if (t + 1 < SNN_T) {
            if (w == 0) {
                tdm_load_x_tile(xbase + (size_t)(t + 1) * SNN_DIN,
                                (unsigned)(uintptr_t)&sXf[1 - cur][0]);
                __builtin_amdgcn_s_wait_tensorcnt(1);   // x(t) complete, x(t+1) in flight
            }
            if (t + 2 < SNN_T)
                __builtin_prefetch(x + (((size_t)(b0 + xr) * SNN_T) + (t + 2)) * SNN_DIN + xc, 0, 1);
        } else {
            if (w == 0)
                __builtin_amdgcn_s_wait_tensorcnt(0);
        }
        __syncthreads();   // barrier A: sXf[cur] visible to all waves

        // ---- Layer 1: cur1 = x_t @ W1^T + b1 ----
        v16h ax0 = fragA_f32(&sXf[cur][lo16 * SNN_DIN + 0  + koff]);
        v16h ax1 = fragA_f32(&sXf[cur][lo16 * SNN_DIN + 32 + koff]);
        v8f acc;
        #pragma unroll
        for (int v = 0; v < 8; v++) acc[v] = b1v;
        acc = __builtin_amdgcn_wmma_f32_16x16x32_f16(false, ax0, false, bw1_0, (short)0, acc, false, false);
        acc = __builtin_amdgcn_wmma_f32_16x16x32_f16(false, ax1, false, bw1_1, (short)0, acc, false, false);

        // Leaky-integrate + subtract-reset + spike; spikes to LDS (f16, [m][h]).
        #pragma unroll
        for (int v = 0; v < 8; v++) {
            float rst = (mem1[v] > SNN_THR) ? SNN_THR : 0.0f;   // reset from PREVIOUS mem
            float m   = SNN_BETA * mem1[v] + acc[v] - rst;
            mem1[v]   = m;
            float s   = (m > SNN_THR) ? 1.0f : 0.0f;
            sS[(v + 8 * hi) * SNN_H + hcol] = (_Float16)s;
        }
        __syncthreads();   // barrier B: spk1 visible

        // ---- Layer 2: cur2 = spk1 @ W2^T + b2 ----
        v16h as0 = fragA_f16(&sS[lo16 * SNN_H + 0  + koff]);
        v16h as1 = fragA_f16(&sS[lo16 * SNN_H + 32 + koff]);
        v8f acc2;
        #pragma unroll
        for (int v = 0; v < 8; v++) acc2[v] = b2v;
        acc2 = __builtin_amdgcn_wmma_f32_16x16x32_f16(false, as0, false, bw2_0, (short)0, acc2, false, false);
        acc2 = __builtin_amdgcn_wmma_f32_16x16x32_f16(false, as1, false, bw2_1, (short)0, acc2, false, false);

        #pragma unroll
        for (int v = 0; v < 8; v++) {
            float rst = (mem2[v] > SNN_THR) ? SNN_THR : 0.0f;
            float m   = SNN_BETA * mem2[v] + acc2[v] - rst;
            mem2[v]   = m;
            float s   = (m > SNN_THR) ? 1.0f : 0.0f;
            ssum[v]  += s;
            outS[(((size_t)(b0 + v + 8 * hi) * SNN_T) + t) * SNN_H + hcol] = s;
        }
        // No trailing barrier needed: next iteration's LDS writes are fenced by barrier A/B.
    }

    // ---- Epilogue: readout_spk = mean_t(spk2) @ Wr^T + br ----
    __syncthreads();   // all layer-2 reads of sS complete before reuse
    #pragma unroll
    for (int v = 0; v < 8; v++)
        sS[(v + 8 * hi) * SNN_H + hcol] = (_Float16)(ssum[v] * (1.0f / SNN_T));
    __syncthreads();

    if (w == 0) {  // one wave does the [16,64]x[64,16] readout tile
        v16h aa0 = fragA_f16(&sS[lo16 * SNN_H + 0  + koff]);
        v16h aa1 = fragA_f16(&sS[lo16 * SNN_H + 32 + koff]);
        v16h bwr0, bwr1;
        const float* wr0 = Wr + lo16 * SNN_H + kb;   // Wr row n, contiguous K
        #pragma unroll
        for (int e = 0; e < 16; e++) {
            bwr0[e] = (_Float16)wr0[e];
            bwr1[e] = (_Float16)wr0[32 + e];
        }
        const float brv = br[lo16];
        v8f accr;
        #pragma unroll
        for (int v = 0; v < 8; v++) accr[v] = brv;
        accr = __builtin_amdgcn_wmma_f32_16x16x32_f16(false, aa0, false, bwr0, (short)0, accr, false, false);
        accr = __builtin_amdgcn_wmma_f32_16x16x32_f16(false, aa1, false, bwr1, (short)0, accr, false, false);
        #pragma unroll
        for (int v = 0; v < 8; v++)
            outR[(b0 + v + 8 * hi) * SNN_DOUT + lo16] = accr[v];
    }
}

extern "C" void kernel_launch(void* const* d_in, const int* in_sizes, int n_in,
                              void* d_out, int out_size, void* d_ws, size_t ws_size,
                              hipStream_t stream) {
    (void)in_sizes; (void)n_in; (void)d_ws; (void)ws_size; (void)out_size;
    const float* x  = (const float*)d_in[0];
    const float* W1 = (const float*)d_in[1];
    const float* b1 = (const float*)d_in[2];
    const float* W2 = (const float*)d_in[3];
    const float* b2 = (const float*)d_in[4];
    const float* Wr = (const float*)d_in[5];
    const float* br = (const float*)d_in[6];
    float* out = (float*)d_out;

    dim3 grid(SNN_B / 16);   // 256 workgroups, each: 16 batch rows
    dim3 block(128);         // 4 waves; wave j owns H columns 16j..16j+15
    snn_small_kernel<<<grid, block, 0, stream>>>(x, W1, b1, W2, b2, Wr, br, out);
}
